// SelectiveSSM_19129784336937
// MI455X (gfx1250) — matrix operand order repaired and checked by reading, not compiled, not run on hardware
//
#include <hip/hip_runtime.h>
#include <hip/hip_bf16.h>
#include <math.h>

#define D_MODEL 1024
#define BATCH   16384

typedef __bf16 bf16t;
typedef __attribute__((ext_vector_type(16))) __bf16 v16bf;
typedef __attribute__((ext_vector_type(8)))  __bf16 v8bf;
typedef __attribute__((ext_vector_type(8)))  float  v8f;
typedef __attribute__((ext_vector_type(4)))  float  v4f;

#define BM 128   // workgroup M tile
#define BN 64    // workgroup N tile
#define BK 32    // K slab (one bf16 WMMA depth)
#define KP 48    // padded LDS K stride (96B: 16B-aligned, conflict-light)

// ---------------------------------------------------------------------------
// Kernel 1: fused LayerNorm + B/C projections + s = sum(Bm*Cm) per row
// One 256-thread block per batch row.
// ---------------------------------------------------------------------------
__global__ __launch_bounds__(256) void ln_s_kernel(
    const float* __restrict__ x, const float* __restrict__ gamma,
    const float* __restrict__ beta,
    const float* __restrict__ Wb, const float* __restrict__ bb,
    const float* __restrict__ Wc, const float* __restrict__ bc,
    float* __restrict__ xn, float* __restrict__ s_out)
{
    const int row  = blockIdx.x;
    const int tid  = threadIdx.x;
    const int lane = tid & 31;
    const int wave = tid >> 5;
    const float* xr = x + (size_t)row * D_MODEL;

    __shared__ float xsh[D_MODEL];   // normalized row for the dot phase
    __shared__ float red[16];
    __shared__ float stats[2];
    __shared__ float dots[32];

    // --- mean / var (wave32 shuffle tree, then 8 wave partials) ---
    float v[4];
    float sum = 0.f, sq = 0.f;
#pragma unroll
    for (int i = 0; i < 4; ++i) {
        v[i] = xr[tid + i * 256];
        sum += v[i];
        sq  += v[i] * v[i];
    }
#pragma unroll
    for (int o = 16; o > 0; o >>= 1) {
        sum += __shfl_xor(sum, o, 32);
        sq  += __shfl_xor(sq,  o, 32);
    }
    if (lane == 0) { red[wave] = sum; red[wave + 8] = sq; }
    __syncthreads();
    if (tid == 0) {
        float ts = 0.f, tq = 0.f;
#pragma unroll
        for (int w = 0; w < 8; ++w) { ts += red[w]; tq += red[w + 8]; }
        float mu  = ts * (1.0f / D_MODEL);
        float var = tq * (1.0f / D_MODEL) - mu * mu;
        stats[0] = mu;
        stats[1] = rsqrtf(var + 1e-5f);
    }
    __syncthreads();
    const float mu = stats[0], rstd = stats[1];

    // --- normalize, write xn, keep row in LDS ---
#pragma unroll
    for (int i = 0; i < 4; ++i) {
        int e = tid + i * 256;
        float xv = (v[i] - mu) * rstd * gamma[e] + beta[e];
        xsh[e] = xv;
        xn[(size_t)row * D_MODEL + e] = xv;
    }
    __syncthreads();

    // --- 32 dots (16 Bm + 16 Cm): wave w handles 4 dots ---
#pragma unroll
    for (int q = 0; q < 4; ++q) {
        int dj = wave * 4 + q;
        const float* Wr = (dj < 16) ? (Wb + (size_t)dj * D_MODEL)
                                    : (Wc + (size_t)(dj - 16) * D_MODEL);
        float acc = 0.f;
#pragma unroll 8
        for (int i = 0; i < 32; ++i) {
            int e = lane + i * 32;
            acc = fmaf(xsh[e], Wr[e], acc);
        }
#pragma unroll
        for (int o = 16; o > 0; o >>= 1) acc += __shfl_xor(acc, o, 32);
        if (lane == 0) dots[dj] = acc;
    }
    __syncthreads();
    if (tid == 0) {
        float s = 0.f;
#pragma unroll
        for (int n = 0; n < 16; ++n)
            s += (dots[n] + bb[n]) * (dots[n + 16] + bc[n]);
        s_out[row] = s;
    }
}

// ---------------------------------------------------------------------------
// Kernels 2/3: split-bf16 WMMA GEMM  C[B,D] = A[B,K] * W[D,K]^T  + epilogue
// EPI==1: y = softplus(C+bdt) * xn * s + xn * D_skip   (writes y)
// EPI==2: out = C + bo + residual                      (writes d_out)
// 256 threads = 8 waves; block tile 128x64; wave tile 32x32 (2x2 frags).
// ---------------------------------------------------------------------------
template <int EPI>
__global__ __launch_bounds__(256) void gemm_fused(
    const float* __restrict__ A, const float* __restrict__ W,
    const float* __restrict__ bias, const float* __restrict__ xnp,
    const float* __restrict__ srow, const float* __restrict__ dskip,
    const float* __restrict__ resid, float* __restrict__ out)
{
    __shared__ __align__(16) bf16t Ah[BM][KP];
    __shared__ __align__(16) bf16t Al[BM][KP];
    __shared__ __align__(16) bf16t Bh[BN][KP];
    __shared__ __align__(16) bf16t Bl[BN][KP];

    const int tid  = threadIdx.x;
    const int lane = tid & 31;
    const int wave = tid >> 5;
    const int wm   = wave >> 1;         // 0..3  (M wave tile)
    const int wn   = wave & 1;          // 0..1  (N wave tile)
    const int l15  = lane & 15;
    const int hi16 = (lane >> 4) & 1;   // upper half-wave?

    const int m0 = blockIdx.y * BM;
    const int n0 = blockIdx.x * BN;

    // staging coordinates (A: 16 floats/thread, B: 8 floats/thread)
    const int arow = tid >> 1;          // 0..127
    const int acol = (tid & 1) * 16;    // 0 | 16
    const int brow = tid >> 2;          // 0..63
    const int bcol = (tid & 3) * 8;     // 0,8,16,24

    v8f acc[2][2] = {};

    for (int k0 = 0; k0 < D_MODEL; k0 += BK) {
        __syncthreads();
        // ---- stage A slab: fp32 -> split bf16 (hi + residual lo) ----
        {
            const float* ga = A + (size_t)(m0 + arow) * D_MODEL + k0 + acol;
#pragma unroll
            for (int j = 0; j < 16; j += 4) {
                v4f vv = *(const v4f*)(ga + j);
#pragma unroll
                for (int e = 0; e < 4; ++e) {
                    float f = vv[e];
                    bf16t h = (bf16t)f;
                    bf16t l = (bf16t)(f - (float)h);
                    Ah[arow][acol + j + e] = h;
                    Al[arow][acol + j + e] = l;
                }
            }
            if (k0 + BK < D_MODEL) __builtin_prefetch(ga + BK, 0, 0);
        }
        // ---- stage W slab ----
        {
            const float* gw = W + (size_t)(n0 + brow) * D_MODEL + k0 + bcol;
#pragma unroll
            for (int j = 0; j < 8; j += 4) {
                v4f vv = *(const v4f*)(gw + j);
#pragma unroll
                for (int e = 0; e < 4; ++e) {
                    float f = vv[e];
                    bf16t h = (bf16t)f;
                    bf16t l = (bf16t)(f - (float)h);
                    Bh[brow][bcol + j + e] = h;
                    Bl[brow][bcol + j + e] = l;
                }
            }
        }
        __syncthreads();

        // ---- build WMMA fragments per the CDNA5 16-bit layouts ----
        v16bf ah[2], al[2], bh[2], bl[2];
#pragma unroll
        for (int mi = 0; mi < 2; ++mi) {
            int r  = wm * 32 + mi * 16 + l15;          // A row (M)
            int o1 = hi16 ? 8 : 0;                     // lanes 16-31: K 8..15 / 24..31
            v8bf c0h = *(const v8bf*)&Ah[r][o1];
            v8bf c1h = *(const v8bf*)&Ah[r][o1 + 16];
            v8bf c0l = *(const v8bf*)&Al[r][o1];
            v8bf c1l = *(const v8bf*)&Al[r][o1 + 16];
            ah[mi] = __builtin_shufflevector(c0h, c1h, 0,1,2,3,4,5,6,7,8,9,10,11,12,13,14,15);
            al[mi] = __builtin_shufflevector(c0l, c1l, 0,1,2,3,4,5,6,7,8,9,10,11,12,13,14,15);
        }
#pragma unroll
        for (int ni = 0; ni < 2; ++ni) {
            int r = wn * 32 + ni * 16 + l15;           // B column (N) = W row
            int o = hi16 ? 8 : 0;                      // lanes 16-31: K 8..15 / 24..31
            v8bf d0h = *(const v8bf*)&Bh[r][o];
            v8bf d1h = *(const v8bf*)&Bh[r][o + 16];
            v8bf d0l = *(const v8bf*)&Bl[r][o];
            v8bf d1l = *(const v8bf*)&Bl[r][o + 16];
            bh[ni] = __builtin_shufflevector(d0h, d1h, 0,1,2,3,4,5,6,7,8,9,10,11,12,13,14,15);
            bl[ni] = __builtin_shufflevector(d0l, d1l, 0,1,2,3,4,5,6,7,8,9,10,11,12,13,14,15);
        }

        // ---- split-bf16 triple product: hi*hi + hi*lo + lo*hi ----
#pragma unroll
        for (int mi = 0; mi < 2; ++mi)
#pragma unroll
            for (int ni = 0; ni < 2; ++ni) {
                acc[mi][ni] = __builtin_amdgcn_wmma_f32_16x16x32_bf16(
                    false, ah[mi], false, bh[ni], (short)0, acc[mi][ni], false, false);
                acc[mi][ni] = __builtin_amdgcn_wmma_f32_16x16x32_bf16(
                    false, ah[mi], false, bl[ni], (short)0, acc[mi][ni], false, false);
                acc[mi][ni] = __builtin_amdgcn_wmma_f32_16x16x32_bf16(
                    false, al[mi], false, bh[ni], (short)0, acc[mi][ni], false, false);
            }
    }

    // ---- fused epilogue via C/D lane mapping: VGPR r -> M=r (+8 for hi half) ----
#pragma unroll
    for (int mi = 0; mi < 2; ++mi)
#pragma unroll
        for (int ni = 0; ni < 2; ++ni)
#pragma unroll
            for (int r = 0; r < 8; ++r) {
                int row = m0 + wm * 32 + mi * 16 + r + hi16 * 8;
                int col = n0 + wn * 32 + ni * 16 + l15;
                size_t idx = (size_t)row * D_MODEL + col;
                float vv = acc[mi][ni][r] + bias[col];
                if (EPI == 1) {
                    float xv = xnp[idx];
                    float sp = (vv > 20.0f) ? vv : log1pf(expf(vv));
                    out[idx] = sp * xv * srow[row] + xv * dskip[col];
                } else {
                    out[idx] = vv + resid[idx];
                }
            }
}

// ---------------------------------------------------------------------------
// Launch
// ---------------------------------------------------------------------------
extern "C" void kernel_launch(void* const* d_in, const int* in_sizes, int n_in,
                              void* d_out, int out_size, void* d_ws, size_t ws_size,
                              hipStream_t stream) {
    const float* x     = (const float*)d_in[0];
    const float* gamma = (const float*)d_in[1];
    const float* beta  = (const float*)d_in[2];
    const float* Wb    = (const float*)d_in[3];
    const float* bb    = (const float*)d_in[4];
    const float* Wc    = (const float*)d_in[5];
    const float* bc    = (const float*)d_in[6];
    const float* Wdt   = (const float*)d_in[7];
    const float* bdt   = (const float*)d_in[8];
    const float* Dskip = (const float*)d_in[9];
    const float* Wo    = (const float*)d_in[10];
    const float* bo    = (const float*)d_in[11];

    // workspace: xn (64MB) | y (64MB) | s (64KB)
    float* xn = (float*)d_ws;
    float* y  = xn + (size_t)BATCH * D_MODEL;
    float* s  = y  + (size_t)BATCH * D_MODEL;

    ln_s_kernel<<<BATCH, 256, 0, stream>>>(x, gamma, beta, Wb, bb, Wc, bc, xn, s);

    dim3 grid(D_MODEL / BN, BATCH / BM);   // (16, 128)
    gemm_fused<1><<<grid, 256, 0, stream>>>(xn, Wdt, bdt, xn, s, Dskip, nullptr, y);
    gemm_fused<2><<<grid, 256, 0, stream>>>(y, Wo, bo, nullptr, nullptr, nullptr, x,
                                            (float*)d_out);
}
